// DimeNetPP_42262478192813
// MI455X (gfx1250) — matrix operand order repaired and checked by reading, not compiled, not run on hardware
//
#include <hip/hip_runtime.h>

// ---------------- types / constants ----------------
typedef __bf16 bf16_t;
typedef bf16_t bf16x16 __attribute__((ext_vector_type(16)));
typedef float  f32x8   __attribute__((ext_vector_type(8)));

#define CUTOFF_F 5.0f

// zeros of spherical Bessel j_l, l=0..6, first 6 roots each
__constant__ float c_zeros[42] = {
  3.14159265358979f, 6.28318530717959f, 9.42477796076938f, 12.56637061435917f, 15.70796326794897f, 18.84955592153876f,
  4.49340945790906f, 7.72525183693771f, 10.90412165942890f, 14.06619391283147f, 17.22075527193077f, 20.37130295928756f,
  5.76345919689455f, 9.09501133047636f, 12.32294097056658f, 15.51460301088675f, 18.68903635536282f, 21.85387422270988f,
  6.98793200050051f, 10.41711854737937f, 13.69802315325025f, 16.92362128521432f, 20.12180617445475f, 23.30424698893965f,
  8.18256145257124f, 11.70490715457039f, 15.03966470761653f, 18.30125595954180f, 21.52541773339994f, 24.72756554783511f,
  9.35581211104275f, 12.96653017277535f, 16.35470963950001f, 19.65315210182131f, 22.90455064790260f, 26.12775013727286f,
  10.51283540809400f, 14.20739245884276f, 17.64797490237652f, 20.98346306895476f, 24.26276818393733f, 27.50786829962061f
};
// Y0[l] = sqrt((2l+1)/(4*pi))
__constant__ float c_y0[7] = {
  0.28209479177388f, 0.48860251190292f, 0.63078313050504f, 0.74635266518023f,
  0.84628437532163f, 0.93560257962738f, 1.01710723627769f
};

__device__ __forceinline__ float siluf(float v) { return v / (1.0f + __expf(-v)); }

__device__ __forceinline__ float jn_dev(int l, float x) {
  float s, c;
  __sincosf(x, &s, &c);
  float inv = 1.0f / x;
  float j0 = s * inv;
  if (l == 0) return j0;
  float jm = j0;
  float jc = (s * inv - c) * inv;
  for (int ll = 1; ll < l; ++ll) {
    float jn1 = (2.0f * ll + 1.0f) * inv * jc - jm;
    jm = jc; jc = jn1;
  }
  return jc;
}

// ---------------- basis kernels ----------------
__global__ __launch_bounds__(64) void k_init_norms(float* __restrict__ norms) {
  int i = threadIdx.x;
  if (i < 42) {
    int l = i / 6;
    norms[i] = 0.126491106406735f / fabsf(jn_dev(l + 1, c_zeros[i])); // sqrt(2/c^3)/|j_{l+1}(z)|
  }
}

__global__ __launch_bounds__(256) void k_basis_edge(
    const float* __restrict__ Dij, const float* __restrict__ norms,
    float* __restrict__ rbf, float* __restrict__ rad, int E)
{
  int e = blockIdx.x * 256 + threadIdx.x;
  if (e >= E) return;
  float xs = Dij[e] * (1.0f / CUTOFF_F);
  float env = 0.0f;
  if (xs < 1.0f) {
    float x2 = xs * xs;
    float x5 = x2 * x2 * xs;
    env = 1.0f / xs + x5 * (-28.0f + xs * (48.0f - 21.0f * xs)); // p=6 envelope
  }
  #pragma unroll
  for (int n = 0; n < 6; ++n)
    rbf[(long)e * 6 + n] = env * __sinf(3.14159265358979f * (float)(n + 1) * xs);
  for (int l = 0; l < 7; ++l)
    for (int n = 0; n < 6; ++n) {
      float xx = xs * c_zeros[l * 6 + n];
      rad[(long)e * 42 + l * 6 + n] = env * norms[l * 6 + n] * jn_dev(l, xx);
    }
}

// ---- pad node_attr [NA,16] f32 -> [NA,32] f32 zero-padded (so node GEMM has K == Kp) ----
__global__ __launch_bounds__(256) void k_pad_node(
    const float* __restrict__ src, float* __restrict__ dst, int NA)
{
  int i = blockIdx.x * 256 + threadIdx.x;
  if (i >= NA * 32) return;
  int a = i >> 5, c = i & 31;
  dst[i] = (c < 16) ? src[a * 16 + c] : 0.0f;
}

// ---- weight prep: f32 [K, Nsrc] (row-major, row_off, ld) -> bf16 [Npad, Kp], zero padded ----
__global__ __launch_bounds__(256) void k_prep_wt(
    const float* __restrict__ W, bf16_t* __restrict__ Wt,
    int K, int Npad, int Nsrc, int ld, int row_off, int Kp)
{
  int idx = blockIdx.x * 256 + threadIdx.x;
  if (idx >= Npad * Kp) return;
  int n = idx / Kp, k = idx - n * Kp;
  float v = (k < K && n < Nsrc) ? W[(long)(row_off + k) * ld + n] : 0.0f;
  Wt[(long)n * Kp + k] = (bf16_t)v;
}

// ---------------- WMMA bf16 GEMM ----------------
// C[M x N] = act(A[M x Kp] @ W + bias) (+ add), C/add leading dim ldc.
// REQUIREMENTS (guaranteed by host): M % 16 == 0, Kp % 32 == 0, N % (16*NB) == 0,
// A row stride == Kp (zero-padded where needed), Wt bf16 [N x Kp] zero-padded.
// Each wave computes a 16 x (16*NB) strip: one A fragment reused across NB WMMAs.
template <int NB>
__global__ __launch_bounds__(256) void k_gemm(
    const float* __restrict__ A, const bf16_t* __restrict__ Wt,
    const float* __restrict__ bias, const float* __restrict__ add,
    float* __restrict__ C, int M, int N, int Kp, int ldc, int act)
{
  const int lane = threadIdx.x & 31;
  const int wv = threadIdx.x >> 5;
  const int tilesNG = N / (16 * NB);
  const long ntiles = (long)(M >> 4) * tilesNG;
  const long tile = (long)blockIdx.x * 8 + wv;
  if (tile >= ntiles) return;              // wave-uniform: EXEC all-ones for WMMA
  const int mt = (int)(tile / tilesNG), ntg = (int)(tile % tilesNG);
  const int m0 = mt << 4, n0 = ntg * (16 * NB);
  const int half = lane >> 4, ml = lane & 15;

  const float* Ar = A + (long)(m0 + ml) * Kp;
  const bf16_t* Br[NB];
  #pragma unroll
  for (int j = 0; j < NB; ++j) Br[j] = Wt + (long)(n0 + j * 16 + ml) * Kp;

  f32x8 acc[NB];
  #pragma unroll
  for (int j = 0; j < NB; ++j) acc[j] = (f32x8){0.f,0.f,0.f,0.f,0.f,0.f,0.f,0.f};

  for (int k0 = 0; k0 < Kp; k0 += 32) {
    // A fragment: elem e -> K = k0 + 16*half + e
    const float4* p0 = (const float4*)(Ar + k0 + (half << 4));
    const float4* p1 = (const float4*)(Ar + k0 + 8 + (half << 4));
    float4 u0 = p0[0], u1 = p0[1], u2 = p1[0], u3 = p1[1];
    bf16x16 a;
    a[0]  = (bf16_t)u0.x; a[1]  = (bf16_t)u0.y; a[2]  = (bf16_t)u0.z; a[3]  = (bf16_t)u0.w;
    a[4]  = (bf16_t)u1.x; a[5]  = (bf16_t)u1.y; a[6]  = (bf16_t)u1.z; a[7]  = (bf16_t)u1.w;
    a[8]  = (bf16_t)u2.x; a[9]  = (bf16_t)u2.y; a[10] = (bf16_t)u2.z; a[11] = (bf16_t)u2.w;
    a[12] = (bf16_t)u3.x; a[13] = (bf16_t)u3.y; a[14] = (bf16_t)u3.z; a[15] = (bf16_t)u3.w;
    __builtin_prefetch(Ar + k0 + 128, 0, 3);
    // B fragments: elem e -> K = k0 + 16*half + e; load all, then issue WMMAs back-to-back
    bf16x16 bfrag[NB];
    #pragma unroll
    for (int j = 0; j < NB; ++j) {
      const uint4* p = (const uint4*)(Br[j] + k0 + (half << 4));
      union { uint4 u[2]; bf16x16 v; } cvt;
      cvt.u[0] = p[0]; cvt.u[1] = p[1];
      bfrag[j] = cvt.v;
    }
    #pragma unroll
    for (int j = 0; j < NB; ++j)
      acc[j] = __builtin_amdgcn_wmma_f32_16x16x32_bf16(false, a, false, bfrag[j],
                                                       (short)0, acc[j], false, false);
  }

  #pragma unroll
  for (int j = 0; j < NB; ++j) {
    const int cn = n0 + j * 16 + ml;
    const float bv = bias ? bias[cn] : 0.0f;
    #pragma unroll
    for (int r = 0; r < 8; ++r) {
      int cm = m0 + r + (half << 3);
      float v = acc[j][r] + bv;
      if (act) v = siluf(v);
      if (add) v += add[(long)cm * ldc + cn];
      C[(long)cm * ldc + cn] = v;
    }
  }
}

// ---------------- elementwise / gather / scatter kernels ----------------
__global__ __launch_bounds__(512) void k_table4(
    const float* __restrict__ emb, const float* __restrict__ mW,
    const float* __restrict__ mb, float* __restrict__ t4)
{
  int t = threadIdx.x >> 7, c = threadIdx.x & 127;
  float s = mb[c];
  for (int k = 0; k < 128; ++k) s += emb[t * 128 + k] * mW[(long)(256 + k) * 128 + c];
  t4[t * 128 + c] = s;
}

__global__ __launch_bounds__(128) void k_edge_assemble(
    const float* __restrict__ H0, const float* __restrict__ H1,
    const float* __restrict__ t4, const float* __restrict__ rbf,
    const float* __restrict__ mW, const int* __restrict__ idi,
    const int* __restrict__ idj, const int* __restrict__ etype,
    float* __restrict__ x)
{
  long e = blockIdx.x;
  int c = threadIdx.x;
  int i = idi[e], j = idj[e], ty = etype[e];
  float v = H0[(long)i * 128 + c] + H1[(long)j * 128 + c] + t4[ty * 128 + c];
  const float* rb = rbf + e * 6;
  #pragma unroll
  for (int n = 0; n < 6; ++n) v += rb[n] * mW[(long)(384 + n) * 128 + c];
  x[e * 128 + c] = siluf(v);
}

__global__ __launch_bounds__(128) void k_rbf_mult(
    const float* __restrict__ rbf, const float* __restrict__ rbf1,
    const float* __restrict__ rbf2, float* __restrict__ buf)
{
  __shared__ float r8[8];
  long e = blockIdx.x;
  int c = threadIdx.x;
  if (c < 8) {
    float s = 0.0f;
    #pragma unroll
    for (int n = 0; n < 6; ++n) s += rbf[e * 6 + n] * rbf1[n * 8 + c];
    r8[c] = s;
  }
  __syncthreads();
  float p = 0.0f;
  #pragma unroll
  for (int j = 0; j < 8; ++j) p += r8[j] * rbf2[j * 128 + c];
  buf[e * 128 + c] *= p;
}

__global__ __launch_bounds__(128) void k_gout(
    const float* __restrict__ rbf, const float* __restrict__ Wrbf,
    const float* __restrict__ xe, float* __restrict__ g)
{
  long e = blockIdx.x;
  int c = threadIdx.x;
  float s = 0.0f;
  #pragma unroll
  for (int n = 0; n < 6; ++n) s += rbf[e * 6 + n] * Wrbf[n * 128 + c];
  g[e * 128 + c] = s * xe[e * 128 + c];
}

__global__ __launch_bounds__(256) void k_segsum(
    const float* __restrict__ src, const int* __restrict__ idx,
    float* __restrict__ dst, int E, int C)
{
  long i = (long)blockIdx.x * 256 + threadIdx.x;
  if (i >= (long)E * C) return;
  long e = i / C; int c = (int)(i - e * C);
  atomicAdd(dst + (long)idx[e] * C + c, src[i]);
}

__global__ __launch_bounds__(256) void k_trip(
    const float* __restrict__ xk, const int* __restrict__ ide, const int* __restrict__ idr,
    const float* __restrict__ rad, const float* __restrict__ ang,
    const float* __restrict__ sbf1, const float* __restrict__ sbf2,
    float* __restrict__ acc, int T)
{
  __shared__ float s8s[4][8];
  int tl = threadIdx.x & 63;
  int ti = threadIdx.x >> 6;
  long t = (long)blockIdx.x * 4 + ti;
  bool valid = (t < T);
  long er = 0, ee = 0;
  if (valid) { er = idr[t]; ee = ide[t]; }
  if (tl < 8) {
    float s = 0.0f;
    if (valid) {
      float ct = __cosf(ang[t]);
      float leg[7];
      leg[0] = 1.0f; leg[1] = ct;
      #pragma unroll
      for (int l = 2; l < 7; ++l)
        leg[l] = ((2.0f * l - 1.0f) * ct * leg[l - 1] - (l - 1.0f) * leg[l - 2]) / (float)l;
      const float* rr = rad + er * 42;
      #pragma unroll
      for (int l = 0; l < 7; ++l) {
        float sv = leg[l] * c_y0[l];
        #pragma unroll
        for (int n = 0; n < 6; ++n)
          s += rr[l * 6 + n] * sv * sbf1[(l * 6 + n) * 8 + tl];
      }
    }
    s8s[ti][tl] = s;
  }
  __syncthreads();
  if (!valid) return;
  float p = 0.0f;
  #pragma unroll
  for (int j = 0; j < 8; ++j) p += s8s[ti][j] * sbf2[j * 64 + tl];
  atomicAdd(acc + er * 64 + tl, xk[ee * 64 + tl] * p);
}

__global__ __launch_bounds__(256) void k_add(
    const float* __restrict__ a, const float* __restrict__ b,
    float* __restrict__ c, long n)
{
  long i = (long)blockIdx.x * 256 + threadIdx.x;
  if (i < n) c[i] = a[i] + b[i];
}

__global__ __launch_bounds__(256) void k_final(
    const float* __restrict__ P, const int* __restrict__ bseg,
    float* __restrict__ out, int NA)
{
  int i = blockIdx.x * 256 + threadIdx.x;
  if (i >= NA * 12) return;
  int a = i / 12, k = i - a * 12;
  atomicAdd(out + bseg[a] * 12 + k, P[a * 16 + k]);   // P has leading dim 16 (padded)
}

// ---------------- host orchestration ----------------
struct Wt { const bf16_t* p; int Kp; };

static void launch_gemm(hipStream_t s, const float* A, Wt w, const float* bias,
                        const float* add, float* C, int M, int N, int ldc, int act) {
  if ((N & 63) == 0) {
    long waves = (long)(M / 16) * (N / 64);
    int blocks = (int)((waves + 7) / 8);
    k_gemm<4><<<blocks, 256, 0, s>>>(A, w.p, bias, add, C, M, N, w.Kp, ldc, act);
  } else {
    long waves = (long)(M / 16) * (N / 16);
    int blocks = (int)((waves + 7) / 8);
    k_gemm<1><<<blocks, 256, 0, s>>>(A, w.p, bias, add, C, M, N, w.Kp, ldc, act);
  }
}

extern "C" void kernel_launch(void* const* d_in, const int* in_sizes, int n_in,
                              void* d_out, int out_size, void* d_ws, size_t ws_size,
                              hipStream_t stream) {
  (void)n_in; (void)ws_size;
  const int E  = in_sizes[1];   // 160000  (multiple of 16)
  const int T  = in_sizes[2];   // 1600000
  const int NA = in_sizes[8];   // 10000   (multiple of 16)

  const float* node_attr = (const float*)d_in[0];
  const float* Dij       = (const float*)d_in[1];
  const float* ang       = (const float*)d_in[2];
  const int*   etype     = (const int*)d_in[3];
  const int*   idi       = (const int*)d_in[4];
  const int*   idj       = (const int*)d_in[5];
  const int*   ide       = (const int*)d_in[6];
  const int*   idr       = (const int*)d_in[7];
  const int*   bseg      = (const int*)d_in[8];
  const float* edge_emb  = (const float*)d_in[9];
  const float* node_W    = (const float*)d_in[10];
  const float* node_b    = (const float*)d_in[11];
  const float* m_W       = (const float*)d_in[12];
  const float* m_b       = (const float*)d_in[13];
  // int block b leaves (insertion order): 0 rbf1,1 rbf2,2 sbf1,3 sbf2,4 Wji,5 bji,6 Wkj,7 bkj,
  //   8 Wdown,9 Wup,10 Wskip,11 bskip, 12..15 res_before{W1,b1,W2,b2}, 16..19 resa0, 20..23 resa1
  auto INT_ = [&](int b, int k) { return (const float*)d_in[14 + 24 * b + k]; };
  // out block o leaves: 0 Wrbf, 1 Wup, 2 dW0, 3 db0, 4 dW1, 5 db1, 6 dW2, 7 db2, 8 Wout
  auto OUT_ = [&](int o, int k) { return (const float*)d_in[110 + 9 * o + k]; };

  // workspace layout (f32)
  float* wsf = (float*)d_ws;
  size_t off = 0;
  auto FA = [&](size_t nf) { float* p = wsf + off; off += nf; return p; };
  float* norms = FA(64);
  float* rbf   = FA((size_t)E * 6);
  float* rad   = FA((size_t)E * 42);
  float* anode = FA((size_t)NA * 32);          // node_attr padded 16 -> 32 cols
  float* h     = FA((size_t)NA * 128);
  float* H0    = FA((size_t)NA * 128);
  float* H1    = FA((size_t)NA * 128);
  float* t4    = FA(512);
  float* x     = FA((size_t)E * 128);
  float* bufA  = FA((size_t)E * 128);
  float* bufB  = FA((size_t)E * 128);
  float* xk64  = FA((size_t)E * 64);
  float* acc64 = FA((size_t)E * 64);
  float* accN  = FA((size_t)NA * 128);
  float* t0    = FA((size_t)NA * 256);
  float* t1    = FA((size_t)NA * 256);
  float* P     = FA((size_t)NA * 16);          // padded: 12 targets in stride-16 rows
  bf16_t* bfpool = (bf16_t*)(wsf + off);

  // ---- weight prep (f32 -> transposed bf16 [Npad, Kp], zero padded) ----
  size_t bfoff = 0;
  auto prep = [&](const float* src, int K, int Npad, int Nsrc, int ld, int row_off) -> Wt {
    int Kp = (K + 31) & ~31;
    bf16_t* dst = bfpool + bfoff;
    bfoff += (size_t)Npad * Kp;
    int total = Npad * Kp;
    k_prep_wt<<<(total + 255) / 256, 256, 0, stream>>>(src, dst, K, Npad, Nsrc, ld, row_off, Kp);
    return Wt{dst, Kp};
  };

  Wt wt_node = prep(node_W, 16, 128, 128, 128, 0);  // Kp=32, rows 16..31 zero
  Wt wt_m0   = prep(m_W, 128, 128, 128, 128, 0);    // rows 0..127   (h_i part)
  Wt wt_m1   = prep(m_W, 128, 128, 128, 128, 128);  // rows 128..255 (h_j part)
  Wt wt_ji[4], wt_kj[4], wt_dn[4], wt_up[4], wt_sk[4], wt_rs[4][3][2];
  for (int b = 0; b < 4; ++b) {
    wt_ji[b] = prep(INT_(b, 4), 128, 128, 128, 128, 0);
    wt_kj[b] = prep(INT_(b, 6), 128, 128, 128, 128, 0);
    wt_dn[b] = prep(INT_(b, 8), 128, 64, 64, 64, 0);
    wt_up[b] = prep(INT_(b, 9), 64, 128, 128, 128, 0);
    wt_sk[b] = prep(INT_(b, 10), 128, 128, 128, 128, 0);
    for (int r = 0; r < 3; ++r) {
      int base = 12 + 4 * r;
      wt_rs[b][r][0] = prep(INT_(b, base + 0), 128, 128, 128, 128, 0);
      wt_rs[b][r][1] = prep(INT_(b, base + 2), 128, 128, 128, 128, 0);
    }
  }
  Wt wt_oup[5], wt_od[5][3], wt_oo[5];
  for (int o = 0; o < 5; ++o) {
    wt_oup[o] = prep(OUT_(o, 1), 128, 256, 256, 256, 0);
    for (int d = 0; d < 3; ++d) wt_od[o][d] = prep(OUT_(o, 2 + 2 * d), 256, 256, 256, 256, 0);
    wt_oo[o] = prep(OUT_(o, 8), 256, 16, 12, 12, 0);  // N padded 12 -> 16
  }

  // ---- pipeline ----
  hipMemsetAsync(d_out, 0, (size_t)out_size * sizeof(float), stream);
  k_init_norms<<<1, 64, 0, stream>>>(norms);
  k_basis_edge<<<(E + 255) / 256, 256, 0, stream>>>(Dij, norms, rbf, rad, E);
  k_pad_node<<<(NA * 32 + 255) / 256, 256, 0, stream>>>(node_attr, anode, NA);

  launch_gemm(stream, anode, wt_node, node_b, nullptr, h, NA, 128, 128, 1);  // h = silu(...)
  launch_gemm(stream, h, wt_m0, nullptr, nullptr, H0, NA, 128, 128, 0);
  launch_gemm(stream, h, wt_m1, nullptr, nullptr, H1, NA, 128, 128, 0);
  k_table4<<<1, 512, 0, stream>>>(edge_emb, m_W, m_b, t4);
  k_edge_assemble<<<E, 128, 0, stream>>>(H0, H1, t4, rbf, m_W, idi, idj, etype, x);

  auto out_block = [&](int o, int first) {
    k_gout<<<E, 128, 0, stream>>>(rbf, OUT_(o, 0), x, bufA);
    hipMemsetAsync(accN, 0, (size_t)NA * 128 * sizeof(float), stream);
    k_segsum<<<(int)(((long)E * 128 + 255) / 256), 256, 0, stream>>>(bufA, idi, accN, E, 128);
    launch_gemm(stream, accN, wt_oup[o], nullptr, nullptr, t0, NA, 256, 256, 0);
    launch_gemm(stream, t0, wt_od[o][0], OUT_(o, 3), nullptr, t1, NA, 256, 256, 1);
    launch_gemm(stream, t1, wt_od[o][1], OUT_(o, 5), nullptr, t0, NA, 256, 256, 1);
    launch_gemm(stream, t0, wt_od[o][2], OUT_(o, 7), nullptr, t1, NA, 256, 256, 1);
    launch_gemm(stream, t1, wt_oo[o], nullptr, first ? nullptr : P, P, NA, 16, 16, 0);
  };

  out_block(0, 1);
  for (int b = 0; b < 4; ++b) {
    launch_gemm(stream, x, wt_ji[b], INT_(b, 5), nullptr, bufA, E, 128, 128, 1); // x_ji
    launch_gemm(stream, x, wt_kj[b], INT_(b, 7), nullptr, bufB, E, 128, 128, 1); // x_kj stage 1
    k_rbf_mult<<<E, 128, 0, stream>>>(rbf, INT_(b, 0), INT_(b, 1), bufB);
    launch_gemm(stream, bufB, wt_dn[b], nullptr, nullptr, xk64, E, 64, 64, 1);   // down
    hipMemsetAsync(acc64, 0, (size_t)E * 64 * sizeof(float), stream);
    k_trip<<<(T + 3) / 4, 256, 0, stream>>>(xk64, ide, idr, rad, ang,
                                            INT_(b, 2), INT_(b, 3), acc64, T);
    launch_gemm(stream, acc64, wt_up[b], nullptr, nullptr, bufB, E, 128, 128, 1); // up
    k_add<<<(int)(((long)E * 128 + 255) / 256), 256, 0, stream>>>(bufA, bufB, bufB, (long)E * 128);
    // res_before
    launch_gemm(stream, bufB, wt_rs[b][0][0], INT_(b, 13), nullptr, bufA, E, 128, 128, 1);
    launch_gemm(stream, bufA, wt_rs[b][0][1], INT_(b, 15), bufB, bufB, E, 128, 128, 1);
    // skip: x = silu(h2@Wskip+bskip) + x
    launch_gemm(stream, bufB, wt_sk[b], INT_(b, 11), x, x, E, 128, 128, 1);
    // res_after x2
    launch_gemm(stream, x, wt_rs[b][1][0], INT_(b, 17), nullptr, bufA, E, 128, 128, 1);
    launch_gemm(stream, bufA, wt_rs[b][1][1], INT_(b, 19), x, x, E, 128, 128, 1);
    launch_gemm(stream, x, wt_rs[b][2][0], INT_(b, 21), nullptr, bufA, E, 128, 128, 1);
    launch_gemm(stream, bufA, wt_rs[b][2][1], INT_(b, 23), x, x, E, 128, 128, 1);
    out_block(b + 1, 0);
  }
  k_final<<<(NA * 12 + 255) / 256, 256, 0, stream>>>(P, bseg, (float*)d_out, NA);
}
